// MOEAdapter_11751030522565
// MI455X (gfx1250) — matrix operand order repaired and checked by reading, not compiled, not run on hardware
//
#include <hip/hip_runtime.h>
#include <hip/hip_bf16.h>

// ---------------- problem constants (match reference) ----------------
#define NTOK 8192
#define DDIM 2048
#define MDIM 512
#define NE   4

// ---------------- tiling ----------------
#define BM        32                 // tokens per block
#define NTHREADS  256                // 8 wave32s
#define XS_STRIDE 40                 // 32 K + 8 pad (bf16) -> 80B row, conflict-free, 16B aligned
#define HS_STRIDE 520                // 512 + 8 pad (bf16) -> 1040B row, conflict-free, 16B aligned

#define WELEMS ((size_t)NE * MDIM * DDIM)   // elements per weight tensor (4,194,304)

typedef __attribute__((ext_vector_type(16))) __bf16 v16bf;
typedef __attribute__((ext_vector_type(8)))  __bf16 v8bf;
typedef __attribute__((ext_vector_type(4)))  __bf16 v4bf;
typedef __attribute__((ext_vector_type(8)))  float  v8f;

#define WMMA_BF16(A, B, C) \
  __builtin_amdgcn_wmma_f32_16x16x32_bf16(false, (A), false, (B), (short)0, (C), false, false)

// ---- A-fragment (16x32 bf16, rows x K) from padded LDS array ----
// ISA 16-bit A layout: lane l holds row (l&15); elems 0-7 = K kh..kh+7,
// elems 8-15 = K kh+16..kh+23, kh = (l>>4)*8.
static __device__ __forceinline__ v16bf lds_a_frag(const __bf16* base, int stride,
                                                   int row0, int kbase, int lane) {
  const int r  = row0 + (lane & 15);
  const int kh = kbase + ((lane >> 4) << 3);
  const __bf16* p = base + r * stride + kh;
  v8bf c0 = *(const v8bf*)(p);
  v8bf c1 = *(const v8bf*)(p + 16);
  return __builtin_shufflevector(c0, c1, 0,1,2,3,4,5,6,7,8,9,10,11,12,13,14,15);
}

// ---- B-fragment (32x16 bf16, K x cols) from pre-split bf16 weights ----
// ISA 16-bit B layout: lane l holds col (l&15); elems 0-7 = K kh..kh+7,
// elems 8-15 = K kh+16..kh+23, kh = (l>>4)*8. Pre-split weights are row-major [outCol][K] bf16,
// so this is two aligned b128 loads per lane, no conversion VALU.
static __device__ __forceinline__ v16bf glb_b_frag_bf(const __bf16* __restrict__ W, int rstride,
                                                      int n0, int kbase, int lane) {
  const int n  = n0 + (lane & 15);
  const int ks = kbase + ((lane >> 4) << 3);
  const __bf16* p = W + (size_t)n * rstride + ks;
  v8bf c0 = *(const v8bf*)(p);
  v8bf c1 = *(const v8bf*)(p + 16);
  return __builtin_shufflevector(c0, c1, 0,1,2,3,4,5,6,7,8,9,10,11,12,13,14,15);
}

// ---------------- kernel 0: zero per-expert counters ----------------
__global__ void moe_init_kernel(int* cnt) {
  if (threadIdx.x < NE) cnt[threadIdx.x] = 0;
}

// ---------------- kernel 1: bucket token ids by expert ----------------
__global__ void moe_bucket_kernel(const int* __restrict__ dom, int* cnt, int* idx) {
  int n = blockIdx.x * blockDim.x + threadIdx.x;
  if (n < NTOK) {
    int e = dom[n] & (NE - 1);
    int p = atomicAdd(&cnt[e], 1);
    idx[e * NTOK + p] = n;
  }
}

// ---------------- kernel 2: one-shot fp32 -> bf16 hi/lo weight split ----------------
// Runs once per launch over W1 and W2 (8.4M elems); removes ~1.6G redundant
// per-block conversion VALU ops from the GEMM hot loop.
__global__ void moe_wcvt_kernel(const float* __restrict__ W1, const float* __restrict__ W2,
                                __bf16* w1h, __bf16* w1l, __bf16* w2h, __bf16* w2l) {
  size_t i = ((size_t)blockIdx.x * blockDim.x + threadIdx.x) * 4;
  const float* src;
  __bf16 *dh, *dl;
  size_t j;
  if (i < WELEMS) { src = W1; dh = w1h; dl = w1l; j = i; }
  else            { src = W2; dh = w2h; dl = w2l; j = i - WELEMS; }
  float4 f = *(const float4*)(src + j);
  float fv[4] = {f.x, f.y, f.z, f.w};
  v4bf hi, lo;
#pragma unroll
  for (int k = 0; k < 4; ++k) {
    __bf16 h = (__bf16)fv[k];
    hi[k] = h;
    lo[k] = (__bf16)(fv[k] - (float)h);
  }
  *(v4bf*)(dh + j) = hi;
  *(v4bf*)(dl + j) = lo;
}

// ---------------- kernel 3: fused per-expert MLP ----------------
__global__ void __launch_bounds__(NTHREADS)
moe_fused_kernel(const float* __restrict__ x,
                 const float* __restrict__ b1, const float* __restrict__ b2,
                 const __bf16* __restrict__ w1h, const __bf16* __restrict__ w1l,
                 const __bf16* __restrict__ w2h, const __bf16* __restrict__ w2l,
                 const int* __restrict__ cnt, const int* __restrict__ idx,
                 float* __restrict__ out) {
  extern __shared__ char smem_raw[];
  __bf16* hs_hi   = (__bf16*)smem_raw;                    // [BM][HS_STRIDE]
  __bf16* hs_lo   = hs_hi + BM * HS_STRIDE;
  __bf16* xs_base = hs_lo + BM * HS_STRIDE;               // 2 ping-pong bufs x (hi+lo)
  int*    tok     = (int*)(xs_base + 4 * BM * XS_STRIDE); // [BM]

  const int e    = blockIdx.y;
  const int cntE = cnt[e];
  const int t0   = blockIdx.x * BM;
  if (t0 >= cntE) return;                                  // uniform exit
  const int valid = (cntE - t0 < BM) ? (cntE - t0) : BM;

  const int tid  = threadIdx.x;
  const int lane = tid & 31;
  const int wave = tid >> 5;

  if (tid < BM) {
    int g = t0 + tid;
    if (g > cntE - 1) g = cntE - 1;                        // padding rows reuse a valid token
    tok[tid] = idx[e * NTOK + g];
  }
  __syncthreads();

  const __bf16* W1h = w1h + (size_t)e * MDIM * DDIM;
  const __bf16* W1l = w1l + (size_t)e * MDIM * DDIM;
  const __bf16* W2h = w2h + (size_t)e * DDIM * MDIM;
  const __bf16* W2l = w2l + (size_t)e * DDIM * MDIM;
  const float*  b1e = b1 + e * MDIM;
  const float*  b2e = b2 + e * DDIM;

  // per-thread staging pointer: row tid>>3, 4 floats at (tid&7)*4
  const float* xrow = x + (size_t)tok[tid >> 3] * DDIM + ((tid & 7) << 2);
  const int    srow = tid >> 3;
  const int    skk  = (tid & 7) << 2;

  // stage one 32-token x 32-K fp32 chunk into LDS buffer `buf` as bf16 hi/lo
  auto stage = [&](int k0, int buf) {
    float4 f = *(const float4*)(xrow + k0);
    float fv[4] = {f.x, f.y, f.z, f.w};
    __bf16* XH = xs_base + buf * (2 * BM * XS_STRIDE);
    __bf16* XL = XH + BM * XS_STRIDE;
#pragma unroll
    for (int j = 0; j < 4; ++j) {
      __bf16 h = (__bf16)fv[j];
      XH[srow * XS_STRIDE + skk + j] = h;
      XL[srow * XS_STRIDE + skk + j] = (__bf16)(fv[j] - (float)h);
    }
  };

  // ================= GEMM1: h = relu(x @ W1^T + b1) =================
  v8f acc[2][4];
#pragma unroll
  for (int r = 0; r < 2; ++r)
#pragma unroll
    for (int c = 0; c < 4; ++c) acc[r][c] = (v8f)0.0f;

  stage(0, 0);
  int cur = 0;
  for (int k0 = 0; k0 < DDIM; k0 += 32) {
    __syncthreads();                                       // buf `cur` ready
    if (k0 + 32 < DDIM) stage(k0 + 32, cur ^ 1);           // overlap next gather

    const __bf16* XH = xs_base + cur * (2 * BM * XS_STRIDE);
    const __bf16* XL = XH + BM * XS_STRIDE;
    v16bf ahi[2], alo[2];
#pragma unroll
    for (int r = 0; r < 2; ++r) {
      ahi[r] = lds_a_frag(XH, XS_STRIDE, r * 16, 0, lane);
      alo[r] = lds_a_frag(XL, XS_STRIDE, r * 16, 0, lane);
    }
#pragma unroll
    for (int c = 0; c < 4; ++c) {
      const int m0 = wave * 64 + c * 16;
      v16bf bhi = glb_b_frag_bf(W1h, DDIM, m0, k0, lane);
      v16bf blo = glb_b_frag_bf(W1l, DDIM, m0, k0, lane);
#pragma unroll
      for (int r = 0; r < 2; ++r) {
        acc[r][c] = WMMA_BF16(ahi[r], bhi, acc[r][c]);     // hi*hi
        acc[r][c] = WMMA_BF16(ahi[r], blo, acc[r][c]);     // hi*lo
        acc[r][c] = WMMA_BF16(alo[r], bhi, acc[r][c]);     // lo*hi
      }
    }
    cur ^= 1;
  }

  // bias + ReLU, split to bf16 hi/lo, park h in LDS
#pragma unroll
  for (int r = 0; r < 2; ++r)
#pragma unroll
    for (int c = 0; c < 4; ++c) {
      const int m    = wave * 64 + c * 16 + (lane & 15);
      const float bs = b1e[m];
#pragma unroll
      for (int v = 0; v < 8; ++v) {
        const int row = r * 16 + v + ((lane >> 4) << 3);
        float hval = acc[r][c][v] + bs;
        hval = fmaxf(hval, 0.0f);
        __bf16 h = (__bf16)hval;
        hs_hi[row * HS_STRIDE + m] = h;
        hs_lo[row * HS_STRIDE + m] = (__bf16)(hval - (float)h);
      }
    }
  __syncthreads();

  // ================= GEMM2: out = x + (h @ W2^T + b2) =================
  for (int cc = 0; cc < 4; ++cc) {
    const int dbase = wave * 256 + cc * 64;
    v8f acc2[2][4];
#pragma unroll
    for (int r = 0; r < 2; ++r)
#pragma unroll
      for (int c = 0; c < 4; ++c) acc2[r][c] = (v8f)0.0f;

    for (int k0 = 0; k0 < MDIM; k0 += 32) {
      v16bf ahi2[2], alo2[2];
#pragma unroll
      for (int r = 0; r < 2; ++r) {
        ahi2[r] = lds_a_frag(hs_hi, HS_STRIDE, r * 16, k0, lane);
        alo2[r] = lds_a_frag(hs_lo, HS_STRIDE, r * 16, k0, lane);
      }
#pragma unroll
      for (int c = 0; c < 4; ++c) {
        const int d0 = dbase + c * 16;
        v16bf bhi = glb_b_frag_bf(W2h, MDIM, d0, k0, lane);
        v16bf blo = glb_b_frag_bf(W2l, MDIM, d0, k0, lane);
#pragma unroll
        for (int r = 0; r < 2; ++r) {
          acc2[r][c] = WMMA_BF16(ahi2[r], bhi, acc2[r][c]);
          acc2[r][c] = WMMA_BF16(ahi2[r], blo, acc2[r][c]);
          acc2[r][c] = WMMA_BF16(alo2[r], bhi, acc2[r][c]);
        }
      }
    }

    // fused epilogue: residual + bias, scattered per-token stores
#pragma unroll
    for (int r = 0; r < 2; ++r)
#pragma unroll
      for (int c = 0; c < 4; ++c) {
        const int d    = dbase + c * 16 + (lane & 15);
        const float bs = b2e[d];
#pragma unroll
        for (int v = 0; v < 8; ++v) {
          const int row = r * 16 + v + ((lane >> 4) << 3);
          if (row < valid) {
            const int t = tok[row];
            const size_t off = (size_t)t * DDIM + d;
            out[off] = x[off] + (acc2[r][c][v] + bs);
          }
        }
      }
  }
}

// ---------------- launcher ----------------
extern "C" void kernel_launch(void* const* d_in, const int* in_sizes, int n_in,
                              void* d_out, int out_size, void* d_ws, size_t ws_size,
                              hipStream_t stream) {
  const float* x   = (const float*)d_in[0];
  const int*   dom = (const int*)d_in[1];
  const float* W1  = (const float*)d_in[2];
  const float* b1  = (const float*)d_in[3];
  const float* W2  = (const float*)d_in[4];
  const float* b2  = (const float*)d_in[5];
  float*       out = (float*)d_out;

  // workspace layout
  char* ws = (char*)d_ws;
  int*  cnt = (int*)ws;                                  // NE counters
  int*  idx = (int*)(ws + 256);                          // NE*NTOK token ids (128KB)
  const size_t WBYTES = WELEMS * 2;                      // 8MB per bf16 tensor
  __bf16* w1h = (__bf16*)(ws + 256 * 1024);
  __bf16* w1l = (__bf16*)(ws + 256 * 1024 + WBYTES);
  __bf16* w2h = (__bf16*)(ws + 256 * 1024 + 2 * WBYTES);
  __bf16* w2l = (__bf16*)(ws + 256 * 1024 + 3 * WBYTES); // total ~33.8MB

  moe_init_kernel<<<1, 64, 0, stream>>>(cnt);
  moe_bucket_kernel<<<NTOK / 256, 256, 0, stream>>>(dom, cnt, idx);

  const int cvtBlocks = (int)((2 * WELEMS / 4) / 256);   // 8192
  moe_wcvt_kernel<<<cvtBlocks, 256, 0, stream>>>(W1, W2, w1h, w1l, w2h, w2l);

  constexpr int SMEM = (2 * BM * HS_STRIDE + 4 * BM * XS_STRIDE) * 2 + BM * 4; // 76928 B
  static bool attr_set = false;                          // host-side only
  if (!attr_set) {
    (void)hipFuncSetAttribute((const void*)moe_fused_kernel,
                              hipFuncAttributeMaxDynamicSharedMemorySize, SMEM);
    attr_set = true;
  }

  dim3 grid(NTOK / BM, NE);
  moe_fused_kernel<<<grid, NTHREADS, SMEM, stream>>>(x, b1, b2, w1h, w1l, w2h, w2l,
                                                     cnt, idx, out);
}
